// QLoraLinear_75067438400019
// MI455X (gfx1250) — matrix operand order repaired and checked by reading, not compiled, not run on hardware
//
#include <hip/hip_runtime.h>
#include <hip/hip_bf16.h>

// ---------------------------------------------------------------------------
// Problem constants (reference: B=4,S=2048,DIN=DOUT=4096,R=16,BLOCK=64,
// SCALING = 32/16 = 2.0)
// ---------------------------------------------------------------------------
#define DIN   4096
#define DOUT  4096
#define RNK   16
#define ABS_STRIDE (DIN / 64)
#define SCALING 2.0f

#define BM 128
#define BN 128
#define BK 64     // two wmma K-steps (K=32 each) per staged tile

typedef __attribute__((ext_vector_type(16))) __bf16 v16bf;
typedef __attribute__((ext_vector_type(8)))  __bf16 v8bf;
typedef __attribute__((ext_vector_type(8)))  float  v8f;
typedef __attribute__((ext_vector_type(4)))  int    v4i;

typedef __attribute__((address_space(1))) v4i as1_v4i;   // global int4
typedef __attribute__((address_space(3))) v4i as3_v4i;   // LDS int4

__device__ __constant__ float c_nf4[16] = {
    -1.0f, -0.6961928009986877f, -0.5250730514526367f, -0.39491748809814453f,
    -0.28444138169288635f, -0.18477343022823334f, -0.09105003625154495f, 0.0f,
    0.07958029955625534f, 0.16093020141124725f, 0.24611230194568634f,
    0.33791524171829224f, 0.44070982933044434f, 0.5626170039176941f,
    0.7229568362236023f, 1.0f};

__device__ __forceinline__ unsigned short f32_to_bf16(float f) {
    unsigned int u = __float_as_uint(f);
    unsigned int r = u + 0x7FFFu + ((u >> 16) & 1u);   // round-to-nearest-even
    return (unsigned short)(r >> 16);
}

// Async global->LDS copy of 16 bytes (CDNA5, ASYNCcnt-tracked). Generic LDS
// pointer low 32 bits == LDS byte offset (flat aperture rule), so the integer
// truncation yields a valid addrspace(3) pointer.
__device__ __forceinline__ void async_copy_b128(const void* g, void* l) {
#if __has_builtin(__builtin_amdgcn_global_load_async_to_lds_b128)
    __builtin_amdgcn_global_load_async_to_lds_b128(
        (as1_v4i*)(uintptr_t)g,
        (as3_v4i*)(unsigned int)(uintptr_t)l,
        /*offset=*/0, /*cpol=*/0);
#else
    *(uint4*)l = *(const uint4*)g;
#endif
}

__device__ __forceinline__ void wait_async0() {
#if __has_builtin(__builtin_amdgcn_s_wait_asynccnt)
    __builtin_amdgcn_s_wait_asynccnt(0);
#else
    asm volatile("s_wait_asynccnt 0x0" ::: "memory");
#endif
}

// ---------------------------------------------------------------------------
// Kernel 1: one-shot NF4 dequant + exact rank-16 LoRA fold into bf16 weights.
//   w_eff[o,k] = NF4[code[o,k]] * absmax[o,k/64] + 2 * sum_r B[o,r]*A[r,k]
// ---------------------------------------------------------------------------
__global__ __launch_bounds__(256) void nf4_dequant_lora_fold(
    const int* __restrict__ codes, const float* __restrict__ absmax,
    const float* __restrict__ Aw, const float* __restrict__ Bw,
    unsigned short* __restrict__ weff)
{
    __shared__ float snf4[16];
    __shared__ float sbw[16];
    const int o   = blockIdx.x;
    const int tid = threadIdx.x;
    if (tid < 16) {
        snf4[tid] = c_nf4[tid];
        sbw[tid]  = Bw[o * RNK + tid];
    }
    __syncthreads();

    float bw[RNK];
#pragma unroll
    for (int r = 0; r < RNK; ++r) bw[r] = sbw[r];

#pragma unroll 4
    for (int t = 0; t < DIN / 256; ++t) {
        const int k    = tid + t * 256;
        const int code = codes[(size_t)o * DIN + k] & 15;
        const float am = absmax[o * ABS_STRIDE + (k >> 6)];
        float lora = 0.0f;
#pragma unroll
        for (int r = 0; r < RNK; ++r)
            lora = fmaf(Aw[(size_t)r * DIN + k], bw[r], lora);
        const float val = snf4[code] * am + SCALING * lora;
        weff[(size_t)o * DIN + k] = f32_to_bf16(val);
    }
}

// ---------------------------------------------------------------------------
// Kernel 2: bf16 WMMA GEMM  out[M,N] = x[M,K] * weff[N,K]^T  (f32 accumulate)
// 256 threads = 8 waves (wave32), 2x4 wave grid, each wave 64x32 via 4x2
// v_wmma_f32_16x16x32_bf16 accumulators. Double-buffered LDS; w tile staged
// with async global->LDS copies.
// ---------------------------------------------------------------------------
__global__ __launch_bounds__(256) void qlora_gemm_bf16(
    const float* __restrict__ x, const unsigned short* __restrict__ weff,
    float* __restrict__ out)
{
    __shared__ __attribute__((aligned(64))) unsigned short sA[2][BM * BK]; // 2x16KB
    __shared__ __attribute__((aligned(64))) unsigned short sB[2][BN * BK]; // 2x16KB

    const int tid  = threadIdx.x;
    const int wave = tid >> 5;
    const int lane = tid & 31;
    const int wm   = wave >> 2;   // 0..1 -> 64 rows
    const int wn   = wave & 3;    // 0..3 -> 32 cols
    const int hlf  = lane >> 4;
    const int l15  = lane & 15;

    const int m0 = blockIdx.y * BM;
    const int n0 = blockIdx.x * BN;

    v8f acc[4][2] = {};

    // Staging assignment: thread -> (row = tid>>1, k-half = (tid&1)*32)
    const int srow = tid >> 1;
    const int sk   = (tid & 1) * 32;

    const float*          gx = x    + (size_t)(m0 + srow) * DIN + sk;
    const unsigned short* gw = weff + (size_t)(n0 + srow) * DIN + sk;

    auto stage = [&](int buf, int kt) {
        // ---- x tile: 32 f32 -> 32 bf16 (8x float4 + 64B of ds stores) ----
        unsigned short* dx = &sA[buf][srow * BK + sk];
#pragma unroll
        for (int v = 0; v < 8; ++v) {
            float4 f = ((const float4*)(gx + kt))[v];
            dx[v * 4 + 0] = f32_to_bf16(f.x);
            dx[v * 4 + 1] = f32_to_bf16(f.y);
            dx[v * 4 + 2] = f32_to_bf16(f.z);
            dx[v * 4 + 3] = f32_to_bf16(f.w);
        }
        // ---- w tile: 64 B per thread, async global->LDS (4x b128) --------
        const unsigned short* src = gw + kt;
        unsigned short*       dw  = &sB[buf][srow * BK + sk];
#pragma unroll
        for (int c = 0; c < 4; ++c)
            async_copy_b128(src + c * 8, dw + c * 8);
    };

    stage(0, 0);
    int buf = 0;

    for (int kt = 0; kt < DIN; kt += BK) {
        wait_async0();        // my async copies for sB[buf] are done
        __syncthreads();      // everyone's tile[buf] is published

        if (kt + BK < DIN) {
            stage(buf ^ 1, kt + BK);
            if (kt + 2 * BK < DIN)
                __builtin_prefetch(gx + kt + 2 * BK, 0, 3);  // global_prefetch_b8
        }

        // ---- math: 2 wmma K-steps over staged BK=64 ----------------------
#pragma unroll
        for (int s = 0; s < 2; ++s) {
            // Preload ALL fragments for this K-step, then 8 back-to-back wmma.
            v16bf bfrag[2];
#pragma unroll
            for (int n = 0; n < 2; ++n) {
                const int col = wn * 32 + n * 16 + l15;
                // B (32x16 bf16): lane(hlf) holds K = s*32+hlf*16+{0..15}
                bfrag[n] = *(const v16bf*)&sB[buf][col * BK + s * 32 + hlf * 16];
            }
            v16bf afrag[4];
#pragma unroll
            for (int m = 0; m < 4; ++m) {
                // A (16x32 bf16): lane holds K = s*32+hlf*16+{0..7} and
                // K = s*32+hlf*16+8+{0..7} -> two 16B chunks.
                const int row = wm * 64 + m * 16 + l15;
                const unsigned short* base = &sA[buf][row * BK + s * 32 + hlf * 16];
                v8bf lo = *(const v8bf*)base;
                v8bf hi = *(const v8bf*)(base + 8);
                afrag[m] = __builtin_shufflevector(
                    lo, hi, 0, 1, 2, 3, 4, 5, 6, 7, 8, 9, 10, 11, 12, 13, 14, 15);
            }
#pragma unroll
            for (int m = 0; m < 4; ++m) {
#pragma unroll
                for (int n = 0; n < 2; ++n) {
                    acc[m][n] = __builtin_amdgcn_wmma_f32_16x16x32_bf16(
                        /*neg_a=*/false, afrag[m], /*neg_b=*/false, bfrag[n],
                        /*c_mod=*/(short)0, acc[m][n],
                        /*reuse_a=*/false, /*reuse_b=*/false);
                }
            }
        }
        buf ^= 1;
    }

    // ---- epilogue: C/D layout (§7.12.2): lanes 0-15 -> M=j, 16-31 -> M=8+j
#pragma unroll
    for (int m = 0; m < 4; ++m) {
        const int rbase = m0 + wm * 64 + m * 16 + hlf * 8;
#pragma unroll
        for (int n = 0; n < 2; ++n) {
            const int col = n0 + wn * 32 + n * 16 + l15;
#pragma unroll
            for (int j = 0; j < 8; ++j) {
                out[(size_t)(rbase + j) * DOUT + col] = acc[m][n][j];
            }
        }
    }
}

// ---------------------------------------------------------------------------
// Launch
// ---------------------------------------------------------------------------
extern "C" void kernel_launch(void* const* d_in, const int* in_sizes, int n_in,
                              void* d_out, int out_size, void* d_ws, size_t ws_size,
                              hipStream_t stream) {
    const float* x      = (const float*)d_in[0];   // [B,S,DIN] f32
    const int*   codes  = (const int*)d_in[1];     // [DOUT,DIN] i32
    const float* absmax = (const float*)d_in[2];   // [DOUT,DIN/64] f32
    const float* Aw     = (const float*)d_in[3];   // [R,DIN] f32
    const float* Bw     = (const float*)d_in[4];   // [DOUT,R] f32
    float*       out    = (float*)d_out;

    const int Mrows = in_sizes[0] / DIN;           // B*S = 8192

    unsigned short* weff = (unsigned short*)d_ws;  // 33.5 MB bf16 scratch

    nf4_dequant_lora_fold<<<DOUT, 256, 0, stream>>>(codes, absmax, Aw, Bw, weff);

    dim3 grid(DOUT / BN, Mrows / BM);
    qlora_gemm_bf16<<<grid, 256, 0, stream>>>(x, weff, out);
}
